// MultiHeadAttention_69861938037416
// MI455X (gfx1250) — matrix operand order repaired and checked by reading, not compiled, not run on hardware
//
#include <hip/hip_runtime.h>
#include <hip/hip_bf16.h>

// ---------- types ----------
typedef __bf16 bf16;
typedef bf16        bf16x16 __attribute__((ext_vector_type(16)));
typedef float       f32x8   __attribute__((ext_vector_type(8)));
typedef unsigned int u32x4  __attribute__((ext_vector_type(4)));
typedef int          i32x4v __attribute__((vector_size(16)));   // matches builtin param

union FragBF {             // 8 VGPRs: 16 bf16 values (A or B fragment of 16x16x32)
    u32x4   q[2];
    bf16x16 v;
};

#define MHA_B  4
#define MHA_S  2048
#define MHA_D  1024
#define MHA_H  16
#define MHA_DK 64

#define AS1 __attribute__((address_space(1)))
#define AS3 __attribute__((address_space(3)))

#if defined(__has_builtin)
#if __has_builtin(__builtin_amdgcn_global_load_async_to_lds_b128)
#define HAVE_ASYNC_LDS 1
#endif
#endif
#ifndef HAVE_ASYNC_LDS
#define HAVE_ASYNC_LDS 0
#endif

__device__ __forceinline__ unsigned short f32_to_bf16(float x) {
    unsigned int u = __float_as_uint(x);
    unsigned int r = (u + 0x7FFFu + ((u >> 16) & 1u)) >> 16;
    return (unsigned short)r;
}

// Stage 16 bytes global -> LDS. Async path uses the gfx1250
// GLOBAL_LOAD_ASYNC_TO_LDS_B128 (ASYNCcnt); fallback round-trips via VGPRs.
__device__ __forceinline__ void stage16(const unsigned short* g, unsigned short* l) {
#if HAVE_ASYNC_LDS
    // flat->LDS: LDS offset is the low 32 bits of the address (aperture in high bits); flat->global is identity
    __builtin_amdgcn_global_load_async_to_lds_b128(
        (AS1 i32x4v*)(unsigned long long)g,
        (AS3 i32x4v*)(unsigned int)(unsigned long long)l, 0, 0);
#else
    *(u32x4*)l = *(const u32x4*)g;
#endif
}

__device__ __forceinline__ void wait_stage() {
#if HAVE_ASYNC_LDS
    asm volatile("s_wait_asynccnt 0" ::: "memory");
#endif
}

// ---------- f32 -> bf16 conversion ----------
__global__ void cvt_f32_bf16_kernel(const float* __restrict__ in,
                                    unsigned short* __restrict__ out, int n) {
    int i = blockIdx.x * blockDim.x + threadIdx.x;
    if (i < n) out[i] = f32_to_bf16(in[i]);
}

// ---------- bf16 GEMM: C[M=8192,N=1024] = A[8192,1024] @ B[1024,1024] + bias ----------
// Block tile 256x64 (8 waves x 32 rows, shared n-slab); B staged in LDS (double buffer).
// mode 0: bf16 out, layout [B,H,S,DK]     (Q, V)
// mode 1: bf16 out, layout [B,H,DK,S]     (K transposed for attention)
// mode 2: f32  out, layout [M,N] (d_out)  (final projection)
__global__ __launch_bounds__(256) void gemm_bf16_kernel(
        const unsigned short* __restrict__ A,
        const unsigned short* __restrict__ Bm,
        const float* __restrict__ bias,
        void* __restrict__ out, int mode) {
    const int tid  = threadIdx.x;
    const int lane = tid & 31;
    const int w    = tid >> 5;
    const int hi   = lane >> 4;          // which 16-lane half
    const int nl   = lane & 15;

    const int n0 = (blockIdx.x & 15) * 64;            // 16 N-slabs, shared by block
    const int mw = (blockIdx.x >> 4) * 256 + w * 32;  // 32 rows per wave

    // LDS: double-buffered 32x64 bf16 B tile (4KB each)
    __shared__ __align__(16) unsigned short sB[2 * 32 * 64];

    // this thread's staging slot: row = tid>>3 (k), 8-col segment = tid&7
    const int srow = tid >> 3, sseg = (tid & 7) * 8;
    unsigned short* ldst = sB + srow * 64 + sseg;
    const unsigned short* gsrc = Bm + (size_t)srow * 1024 + n0 + sseg;

    f32x8 acc[2][4] = {};
    const unsigned short* Arow0 = A + (size_t)(mw + nl) * 1024;
    const unsigned short* Arow1 = Arow0 + (size_t)16 * 1024;

    stage16(gsrc, ldst);                               // prologue: stage k0=0 into buf 0

    for (int k0 = 0; k0 < 1024; k0 += 32) {
        const int buf = (k0 >> 5) & 1;
        wait_stage();
        __syncthreads();                               // staged tile visible to all waves
        if (k0 + 32 < 1024)
            stage16(gsrc + (size_t)(k0 + 32) * 1024, ldst + (buf ^ 1) * 2048);

        FragBF a0, a1;
        a0.q[0] = *(const u32x4*)(Arow0 + k0 + hi * 8);
        a0.q[1] = *(const u32x4*)(Arow0 + k0 + 16 + hi * 8);
        a1.q[0] = *(const u32x4*)(Arow1 + k0 + hi * 8);
        a1.q[1] = *(const u32x4*)(Arow1 + k0 + 16 + hi * 8);

        const unsigned short* bcol = sB + buf * 2048 + nl;
#pragma unroll
        for (int j = 0; j < 4; ++j) {
            FragBF b;                                  // lane holds B[k0 + table-order k, n0+16j+nl]
#pragma unroll
            for (int s = 0; s < 16; ++s) {
                const int kk = (s < 8) ? (hi * 8 + s) : (16 + hi * 8 + (s - 8));
                b.v[s] = __builtin_bit_cast(bf16, bcol[kk * 64 + 16 * j]);
            }
            acc[0][j] = __builtin_amdgcn_wmma_f32_16x16x32_bf16(
                false, a0.v, false, b.v, (short)0, acc[0][j], false, false);
            acc[1][j] = __builtin_amdgcn_wmma_f32_16x16x32_bf16(
                false, a1.v, false, b.v, (short)0, acc[1][j], false, false);
        }
        __syncthreads();                               // all reads of buf done before restage
    }

#pragma unroll
    for (int t = 0; t < 2; ++t) {
#pragma unroll
        for (int j = 0; j < 4; ++j) {
#pragma unroll
            for (int i = 0; i < 8; ++i) {
                const int m = mw + t * 16 + i + 8 * hi;   // C layout: M = vgpr + 8*half
                const int n = n0 + 16 * j + nl;           //           N = lane&15
                const float val = acc[t][j][i] + bias[n];
                if (mode == 2) {
                    ((float*)out)[(size_t)m * 1024 + n] = val;
                } else {
                    const int b = m >> 11, s = m & 2047;
                    const int h = n >> 6,  dk = n & 63;
                    const size_t idx = (mode == 0)
                        ? ((((size_t)b * MHA_H + h) * MHA_S + s) * MHA_DK + dk)
                        : ((((size_t)b * MHA_H + h) * MHA_DK + dk) * MHA_S + s);
                    ((unsigned short*)out)[idx] = f32_to_bf16(val);
                }
            }
        }
    }
}

// ---------- flash attention: 4 waves/block, 16 q-rows/wave ----------
// Q,V: [B,H,S,DK] bf16 ; Kt: [B,H,DK,S] bf16 ; ctx out: [B,S,H*DK] bf16
__global__ __launch_bounds__(128) void attn_kernel(
        const unsigned short* __restrict__ Q,
        const unsigned short* __restrict__ Kt,
        const unsigned short* __restrict__ V,
        unsigned short* __restrict__ ctx) {
    const int lane = threadIdx.x & 31;
    const int w    = threadIdx.x >> 5;
    const int hi   = lane >> 4;
    const int nl   = lane & 15;

    const int blk = blockIdx.x;          // (bh * 32 + qblock)
    const int qb  = blk & 31;
    const int bh  = blk >> 5;            // 0..63
    const int q0  = qb * 64 + w * 16;

    const unsigned short* Qbh  = Q  + (size_t)bh * MHA_S * MHA_DK;
    const unsigned short* Ktbh = Kt + (size_t)bh * MHA_DK * MHA_S;
    const unsigned short* Vbh  = V  + (size_t)bh * MHA_S * MHA_DK;

    __shared__ __align__(16) unsigned short smem[4 * 16 * 32];
    unsigned short* sm = smem + w * 512;

    // Q A-fragments (held for the whole block): 16 rows x 64 dims = 2 fragments
    FragBF aq[2];
    const unsigned short* qrow = Qbh + (size_t)(q0 + nl) * MHA_DK;
#pragma unroll
    for (int c = 0; c < 2; ++c) {
        aq[c].q[0] = *(const u32x4*)(qrow + c * 32 + hi * 8);
        aq[c].q[1] = *(const u32x4*)(qrow + c * 32 + 16 + hi * 8);
    }

    float mrow[8], ssum[8];
#pragma unroll
    for (int i = 0; i < 8; ++i) { mrow[i] = -1e30f; ssum[i] = 0.0f; }
    f32x8 o[4] = {};

    const int xmask[4] = {1, 2, 4, 8};

    for (int kb = 0; kb < MHA_S; kb += 32) {
        if (kb + 32 < MHA_S)
            __builtin_prefetch(Vbh + (size_t)(kb + 32 + lane) * MHA_DK, 0, 1);

        // scores = (Q K^T) for 32 keys, as two 16x16 tiles; K-dim = DK = 64 (2 chunks)
        f32x8 s0 = {}, s1 = {};
#pragma unroll
        for (int c = 0; c < 2; ++c) {
            const unsigned short* kp = Ktbh + (size_t)(c * 32) * MHA_S + kb + nl;
            FragBF bk;
#pragma unroll
            for (int s = 0; s < 16; ++s) {
                const int kk = (s < 8) ? (hi * 8 + s) : (16 + hi * 8 + (s - 8));
                bk.v[s] = __builtin_bit_cast(bf16, kp[(size_t)kk * MHA_S]);
            }
            s0 = __builtin_amdgcn_wmma_f32_16x16x32_bf16(
                false, aq[c].v, false, bk.v, (short)0, s0, false, false);
#pragma unroll
            for (int s = 0; s < 16; ++s) {
                const int kk = (s < 8) ? (hi * 8 + s) : (16 + hi * 8 + (s - 8));
                bk.v[s] = __builtin_bit_cast(bf16, kp[(size_t)kk * MHA_S + 16]);
            }
            s1 = __builtin_amdgcn_wmma_f32_16x16x32_bf16(
                false, aq[c].v, false, bk.v, (short)0, s1, false, false);
        }

        // online softmax update (row r = i + 8*hi lives in one VGPR across 16 lanes)
        float p0[8], p1[8];
#pragma unroll
        for (int i = 0; i < 8; ++i) {
            float a0 = s0[i] * 0.125f;                 // 1/sqrt(DK)
            float a1 = s1[i] * 0.125f;
            float t = fmaxf(a0, a1);
#pragma unroll
            for (int x = 0; x < 4; ++x) t = fmaxf(t, __shfl_xor(t, xmask[x], 32));
            const float nm   = fmaxf(mrow[i], t);
            const float corr = __expf(mrow[i] - nm);
            p0[i] = __expf(a0 - nm);
            p1[i] = __expf(a1 - nm);
            float ls = p0[i] + p1[i];
#pragma unroll
            for (int x = 0; x < 4; ++x) ls += __shfl_xor(ls, xmask[x], 32);
            ssum[i] = ssum[i] * corr + ls;
            mrow[i] = nm;
#pragma unroll
            for (int j = 0; j < 4; ++j) o[j][i] *= corr;
        }

        // C-layout -> A-layout round trip through LDS (bf16, row-major 16x32)
#pragma unroll
        for (int i = 0; i < 8; ++i) {
            sm[(i + 8 * hi) * 32 + nl]      = f32_to_bf16(p0[i]);
            sm[(i + 8 * hi) * 32 + 16 + nl] = f32_to_bf16(p1[i]);
        }
        asm volatile("s_wait_dscnt 0" ::: "memory");

        FragBF pa;
        {
            const unsigned short* pr = sm + nl * 32 + hi * 8;
            pa.q[0] = *(const u32x4*)(pr);
            pa.q[1] = *(const u32x4*)(pr + 16);
        }
        asm volatile("" ::: "memory");   // keep next iteration's stores after these loads

        // O += P @ V (K-dim = 32 keys), 4 N-tiles cover DK=64
#pragma unroll
        for (int j = 0; j < 4; ++j) {
            const unsigned short* vp = Vbh + (size_t)kb * MHA_DK + 16 * j + nl;
            FragBF bv;
#pragma unroll
            for (int s = 0; s < 16; ++s) {
                const int kk = (s < 8) ? (hi * 8 + s) : (16 + hi * 8 + (s - 8));
                bv.v[s] = __builtin_bit_cast(bf16, vp[(size_t)kk * MHA_DK]);
            }
            o[j] = __builtin_amdgcn_wmma_f32_16x16x32_bf16(
                false, pa.v, false, bv.v, (short)0, o[j], false, false);
        }
    }

    // normalize and write context in [B,S,H*DK] (natural layout for output GEMM)
    const int b = bh >> 4, h = bh & 15;
#pragma unroll
    for (int i = 0; i < 8; ++i) {
        const float inv = 1.0f / ssum[i];
        const int s = q0 + i + 8 * hi;
#pragma unroll
        for (int j = 0; j < 4; ++j) {
            const int n = h * MHA_DK + 16 * j + nl;
            ctx[((size_t)b * MHA_S + s) * (MHA_H * MHA_DK) + n] = f32_to_bf16(o[j][i] * inv);
        }
    }
}

// ---------- launcher ----------
extern "C" void kernel_launch(void* const* d_in, const int* in_sizes, int n_in,
                              void* d_out, int out_size, void* d_ws, size_t ws_size,
                              hipStream_t stream) {
    (void)in_sizes; (void)n_in; (void)out_size; (void)ws_size;

    const float* q_in = (const float*)d_in[0];
    const float* k_in = (const float*)d_in[1];
    const float* v_in = (const float*)d_in[2];
    const float* Wq = (const float*)d_in[3];  const float* bq = (const float*)d_in[4];
    const float* Wk = (const float*)d_in[5];  const float* bk = (const float*)d_in[6];
    const float* Wv = (const float*)d_in[7];  const float* bv = (const float*)d_in[8];
    const float* Wo = (const float*)d_in[9];  const float* bo = (const float*)d_in[10];

    const size_t ACT = (size_t)MHA_B * MHA_S * MHA_D;        // 8,388,608
    const size_t WEL = (size_t)MHA_D * MHA_D;                // 1,048,576
    unsigned short* ws = (unsigned short*)d_ws;

    unsigned short* qa  = ws;                 // bf16 activations
    unsigned short* ka  = qa  + ACT;
    unsigned short* va  = ka  + ACT;
    unsigned short* wqb = va  + ACT;          // bf16 weights
    unsigned short* wkb = wqb + WEL;
    unsigned short* wvb = wkb + WEL;
    unsigned short* wob = wvb + WEL;
    unsigned short* Qh  = wob + WEL;          // [B,H,S,DK]
    unsigned short* Kt  = Qh  + ACT;          // [B,H,DK,S]
    unsigned short* Vh  = Kt  + ACT;          // [B,H,S,DK]
    unsigned short* ctx = Vh  + ACT;          // [B,S,H*DK]

    const int T = 256;
    // f32 -> bf16
    cvt_f32_bf16_kernel<<<(int)((ACT + T - 1) / T), T, 0, stream>>>(q_in, qa, (int)ACT);
    cvt_f32_bf16_kernel<<<(int)((ACT + T - 1) / T), T, 0, stream>>>(k_in, ka, (int)ACT);
    cvt_f32_bf16_kernel<<<(int)((ACT + T - 1) / T), T, 0, stream>>>(v_in, va, (int)ACT);
    cvt_f32_bf16_kernel<<<(int)((WEL + T - 1) / T), T, 0, stream>>>(Wq, wqb, (int)WEL);
    cvt_f32_bf16_kernel<<<(int)((WEL + T - 1) / T), T, 0, stream>>>(Wk, wkb, (int)WEL);
    cvt_f32_bf16_kernel<<<(int)((WEL + T - 1) / T), T, 0, stream>>>(Wv, wvb, (int)WEL);
    cvt_f32_bf16_kernel<<<(int)((WEL + T - 1) / T), T, 0, stream>>>(Wo, wob, (int)WEL);

    // projections: block tile 256x64 -> (8192/256)*(1024/64) = 512 blocks
    gemm_bf16_kernel<<<512, 256, 0, stream>>>(qa, wqb, bq, Qh,  0);
    gemm_bf16_kernel<<<512, 256, 0, stream>>>(ka, wkb, bk, Kt,  1);
    gemm_bf16_kernel<<<512, 256, 0, stream>>>(va, wvb, bv, Vh,  0);

    // attention: B*H * (S/64) blocks of 4 waves
    attn_kernel<<<MHA_B * MHA_H * (MHA_S / 64), 128, 0, stream>>>(Qh, Kt, Vh, ctx);

    // output projection -> f32 d_out
    gemm_bf16_kernel<<<512, 256, 0, stream>>>(ctx, wob, bo, d_out, 2);
}
